// FP8RotateQuantFA_30013231464947
// MI455X (gfx1250) — 16-bit and 8-bit matrix operand order repaired and checked by reading, compiled, not run on hardware
//
#include <hip/hip_runtime.h>
#include <cstdint>
#include <cstddef>

typedef __attribute__((ext_vector_type(2)))  int      v2i;
typedef __attribute__((ext_vector_type(4)))  int      v4i;
typedef __attribute__((ext_vector_type(16))) int      v16i;
typedef __attribute__((ext_vector_type(8)))  float    v8f;
typedef __attribute__((ext_vector_type(8)))  _Float16 v8h;
typedef __attribute__((ext_vector_type(16))) _Float16 v16h;

#if __has_builtin(__builtin_amdgcn_cvt_pk_fp8_f32)
#define HW_FP8_ENC 1
#endif
#if __has_builtin(__builtin_amdgcn_cvt_f32_fp8)
#define HW_FP8_DEC 1
#endif

__device__ inline float fast_exp2(float x) {
#if __has_builtin(__builtin_amdgcn_exp2f)
  return __builtin_amdgcn_exp2f(x);
#else
  return exp2f(x);
#endif
}

__device__ inline float dec_e4m3_sw(uint32_t b) {
  uint32_t e = (b >> 3) & 0xF, m = b & 7;
  float mag = e ? ldexpf((float)(8u + m), (int)e - 10) : ldexpf((float)m, -9);
  return (b & 0x80u) ? -mag : mag;
}

__device__ inline uint32_t enc_e4m3_sw(float x) {
  uint32_t s = (__float_as_uint(x) >> 31) << 7;
  float a = fabsf(x);
  if (a < 0.015625f) {
    uint32_t mi = (uint32_t)rintf(a * 512.0f);
    return (mi >= 8u) ? (s | 8u) : (s | mi);
  }
  int ex; float fr = frexpf(a, &ex);
  int E = ex - 1;
  uint32_t mi = (uint32_t)rintf(fr * 16.0f - 8.0f);
  if (mi == 8u) { mi = 0u; ++E; }
  if (E > 8) { E = 8; mi = 6u; }
  if (E == 8 && mi == 7u) mi = 6u;
  return s | ((uint32_t)(E + 7) << 3) | mi;
}

__device__ inline uint8_t enc_e4m3(float x) {
#if defined(HW_FP8_ENC)
  return (uint8_t)(__builtin_amdgcn_cvt_pk_fp8_f32(x, x, 0, false) & 0xFF);
#else
  return (uint8_t)enc_e4m3_sw(x);
#endif
}

__device__ inline void dec4_e4m3(int word, float f[4]) {
#if defined(HW_FP8_DEC)
  f[0] = __builtin_amdgcn_cvt_f32_fp8(word, 0);
  f[1] = __builtin_amdgcn_cvt_f32_fp8(word, 1);
  f[2] = __builtin_amdgcn_cvt_f32_fp8(word, 2);
  f[3] = __builtin_amdgcn_cvt_f32_fp8(word, 3);
#else
  f[0] = dec_e4m3_sw(((uint32_t)word >> 0)  & 0xFFu);
  f[1] = dec_e4m3_sw(((uint32_t)word >> 8)  & 0xFFu);
  f[2] = dec_e4m3_sw(((uint32_t)word >> 16) & 0xFFu);
  f[3] = dec_e4m3_sw(((uint32_t)word >> 24) & 0xFFu);
#endif
}

template <int CTRL>
__device__ inline float dpp_mov(float x) {
  return __int_as_float(__builtin_amdgcn_update_dpp(
      0, __float_as_int(x), CTRL, 0xF, 0xF, true));
}
__device__ inline float row_max16(float v) {
  v = fmaxf(v, dpp_mov<0xB1>(v));
  v = fmaxf(v, dpp_mov<0x4E>(v));
  v = fmaxf(v, dpp_mov<0x141>(v));
  v = fmaxf(v, dpp_mov<0x140>(v));
  return v;
}
__device__ inline float row_sum16(float v) {
  v += dpp_mov<0xB1>(v);
  v += dpp_mov<0x4E>(v);
  v += dpp_mov<0x141>(v);
  v += dpp_mov<0x140>(v);
  return v;
}

template <bool ROT>
__global__ __launch_bounds__(256) void rotquant_kernel(
    const float* __restrict__ in, const float* __restrict__ rot,
    uint8_t* __restrict__ out8, float* __restrict__ scales,
    int S, int BS, int nBlocks) {
  constexpr int D = 128;
  __shared__ float ldsR[ROT ? D * D : 8];

  const int tid  = threadIdx.x;
  const int head = blockIdx.y;
  const int bx   = blockIdx.x;
  const size_t base = ((size_t)head * S + (size_t)bx * BS) * D;

  if (ROT) {
    for (int i = tid * 4; i < D * D; i += 256 * 4)
      *reinterpret_cast<float4*>(&ldsR[i]) =
          *reinterpret_cast<const float4*>(rot + i);
    __syncthreads();
  }

  auto value_at = [&](int idx) -> float {
    if (ROT) {
      const int r = idx >> 7, d = idx & 127;
      const float* inrow = in + base + (size_t)r * D;
      float acc = 0.f;
      #pragma unroll 8
      for (int k = 0; k < D; k += 4) {
        float4 iv = *reinterpret_cast<const float4*>(inrow + k);
        acc = fmaf(iv.x, ldsR[(k + 0) * D + d], acc);
        acc = fmaf(iv.y, ldsR[(k + 1) * D + d], acc);
        acc = fmaf(iv.z, ldsR[(k + 2) * D + d], acc);
        acc = fmaf(iv.w, ldsR[(k + 3) * D + d], acc);
      }
      return acc;
    } else {
      return in[base + idx];
    }
  };

  const int total = BS * D;
  float amax = 0.f;
  for (int idx = tid; idx < total; idx += 256)
    amax = fmaxf(amax, fabsf(value_at(idx)));

  #pragma unroll
  for (int m = 1; m < 32; m <<= 1)
    amax = fmaxf(amax, __shfl_xor(amax, m, 32));

  float saved = (tid < 8) ? ldsR[tid] : 0.f;
  __syncthreads();
  if ((tid & 31) == 0) ldsR[tid >> 5] = amax;
  __syncthreads();
  float bm = ldsR[0];
  #pragma unroll
  for (int w = 1; w < 8; ++w) bm = fmaxf(bm, ldsR[w]);
  __syncthreads();
  if (tid < 8) ldsR[tid] = saved;
  __syncthreads();

  const float scale = fmaxf(bm, 1e-12f) * (1.0f / 448.0f);
  const float inv   = 1.0f / scale;
  if (tid == 0) scales[head * nBlocks + bx] = scale;

  for (int idx = tid; idx < total; idx += 256)
    out8[base + idx] = enc_e4m3(value_at(idx) * inv);
}

__global__ __launch_bounds__(256) void fa_fp8_kernel(
    const uint8_t* __restrict__ q8, const uint8_t* __restrict__ k8,
    const uint8_t* __restrict__ v8, const float* __restrict__ qs,
    const float* __restrict__ ks, const float* __restrict__ vs,
    float* __restrict__ out) {
  constexpr int S = 2048, D = 128;
  constexpr int PSTR = 80;
  __shared__ _Float16 Plds[8 * 16 * PSTR];
  __shared__ _Float16 Vlds[2][128 * PSTR];

  const int tid  = threadIdx.x;
  const int wave = tid >> 5, lane = tid & 31;
  const int n = lane & 15, h = lane >> 4;
  const int head = blockIdx.y, kvh = head >> 1;
  const int qb = blockIdx.x;

  const uint8_t* vbase = v8 + (size_t)kvh * S * D;
  const uint8_t* kbase = k8 + (size_t)kvh * S * D;

  const uint8_t* qrow =
      q8 + ((size_t)head * S + (size_t)qb * 128 + wave * 16 + n) * D;
  v16i A;
  #pragma unroll
  for (int b = 0; b < 2; ++b)
    #pragma unroll
    for (int p = 0; p < 4; ++p) {
      v2i t = *reinterpret_cast<const v2i*>(qrow + 64 * b + 16 * p + 8 * h);
      A[b * 8 + p * 2 + 0] = t.x;
      A[b * 8 + p * 2 + 1] = t.y;
    }
  const float sq = qs[head * 16 + qb];

  float m_run[8], l_run[8], alpha[8];
  v8f acc[8];
  #pragma unroll
  for (int j = 0; j < 8; ++j) { m_run[j] = -3.0e38f; l_run[j] = 0.f; }
  #pragma unroll
  for (int dt = 0; dt < 8; ++dt)
    #pragma unroll
    for (int j = 0; j < 8; ++j) acc[dt][j] = 0.f;

  _Float16* pw = Plds + wave * 16 * PSTR;

  const int kk = tid >> 2, dbase = (tid & 3) * 32;
  auto loadV = [&](int kb, v4i& a, v4i& b) {
    const uint8_t* vrow = vbase + (size_t)(kb + kk) * D + dbase;
    a = *reinterpret_cast<const v4i*>(vrow);
    b = *reinterpret_cast<const v4i*>(vrow + 16);
  };
  auto storeV = [&](int buf, v4i a, v4i b) {
    #pragma unroll
    for (int w = 0; w < 8; ++w) {
      int word = (w < 4) ? a[w] : b[w - 4];
      float f[4];
      dec4_e4m3(word, f);
      #pragma unroll
      for (int e = 0; e < 4; ++e)
        Vlds[buf][(dbase + 4 * w + e) * PSTR + kk] = (_Float16)f[e];
    }
  };

  {
    v4i c0, c1;
    loadV(0, c0, c1);
    storeV(0, c0, c1);
  }
  __syncthreads();

  constexpr float LOG2E = 1.44269504088896340736f;

  for (int kb = 0; kb < S; kb += 64) {
    const int buf = (kb >> 6) & 1;
    const bool more = (kb + 64) < S;

    v4i n0 = {}, n1 = {};
    if (more) loadV(kb + 64, n0, n1);
    if (kb + 128 < S)
      __builtin_prefetch(vbase + (size_t)(kb + 128 + kk) * D + dbase, 0, 1);
    if (more)
      __builtin_prefetch(kbase + (size_t)(kb + 64 + (tid & 63)) * D, 0, 1);

    const float sk = ks[kvh * 8 + (kb >> 8)];
    const float sv = vs[kvh * 8 + (kb >> 8)];
    const float sc = sq * sk * (0.08838834764831845f * LOG2E);

    v8f s[4];
    v8f zero = {};
    #pragma unroll
    for (int t = 0; t < 4; ++t) {
      const uint8_t* krow = kbase + (size_t)(kb + 16 * t + n) * D;
      v16i B;
      #pragma unroll
      for (int g = 0; g < 4; ++g) {
        v2i w0 = *reinterpret_cast<const v2i*>(krow + 32 * g + 8 * h);
        v2i w1 = *reinterpret_cast<const v2i*>(krow + 32 * g + 16 + 8 * h);
        B[4 * g + 0] = w0.x; B[4 * g + 1] = w0.y;
        B[4 * g + 2] = w1.x; B[4 * g + 3] = w1.y;
      }
      s[t] = __builtin_amdgcn_wmma_f32_16x16x128_fp8_fp8(A, B, (short)0, zero,
                                                         false, false);
    }
    #pragma unroll
    for (int t = 0; t < 4; ++t)
      #pragma unroll
      for (int j = 0; j < 8; ++j) s[t][j] *= sc;

    #pragma unroll
    for (int j = 0; j < 8; ++j) {
      float rm = fmaxf(fmaxf(s[0][j], s[1][j]), fmaxf(s[2][j], s[3][j]));
      rm = row_max16(rm);
      const float mnew = fmaxf(m_run[j], rm);
      const float al = fast_exp2(m_run[j] - mnew);
      m_run[j] = mnew; alpha[j] = al;
      #pragma unroll
      for (int t = 0; t < 4; ++t) s[t][j] = fast_exp2(s[t][j] - mnew);
      float rs = s[0][j] + s[1][j] + s[2][j] + s[3][j];
      rs = row_sum16(rs);
      l_run[j] = l_run[j] * al + rs;
    }
    #pragma unroll
    for (int dt = 0; dt < 8; ++dt)
      #pragma unroll
      for (int j = 0; j < 8; ++j) acc[dt][j] *= alpha[j];

    #pragma unroll
    for (int j = 0; j < 8; ++j)
      #pragma unroll
      for (int t = 0; t < 4; ++t)
        pw[(j + 8 * h) * PSTR + 16 * t + n] = (_Float16)(s[t][j] * sv);

    #pragma unroll
    for (int s2 = 0; s2 < 2; ++s2) {
      const _Float16* pa = pw + n * PSTR + 32 * s2 + 8 * h;
      v8h plo = *reinterpret_cast<const v8h*>(pa);
      v8h phi = *reinterpret_cast<const v8h*>(pa + 16);
      v16h P16 = __builtin_shufflevector(plo, phi, 0, 1, 2, 3, 4, 5, 6, 7, 8,
                                         9, 10, 11, 12, 13, 14, 15);
      #pragma unroll
      for (int dt = 0; dt < 8; ++dt) {
        const _Float16* vb =
            &Vlds[buf][(dt * 16 + n) * PSTR + 32 * s2 + 8 * h];
        v8h vlo = *reinterpret_cast<const v8h*>(vb);
        v8h vhi = *reinterpret_cast<const v8h*>(vb + 16);
        v16h V16 = __builtin_shufflevector(vlo, vhi, 0, 1, 2, 3, 4, 5, 6, 7,
                                           8, 9, 10, 11, 12, 13, 14, 15);
        acc[dt] = __builtin_amdgcn_wmma_f32_16x16x32_f16(
            false, P16, false, V16, (short)0, acc[dt], false, false);
      }
    }

    if (more) storeV(buf ^ 1, n0, n1);
    __syncthreads();
  }

  #pragma unroll
  for (int j = 0; j < 8; ++j) {
    const float invl = 1.0f / l_run[j];
    const size_t row = (size_t)head * S + (size_t)qb * 128 + wave * 16 + j + 8 * h;
    #pragma unroll
    for (int dt = 0; dt < 8; ++dt)
      out[row * D + dt * 16 + n] = acc[dt][j] * invl;
  }
}

extern "C" void kernel_launch(void* const* d_in, const int* in_sizes, int n_in,
                              void* d_out, int out_size, void* d_ws,
                              size_t ws_size, hipStream_t stream) {
  (void)in_sizes; (void)n_in; (void)out_size; (void)ws_size;
  const float* q   = (const float*)d_in[0];
  const float* k   = (const float*)d_in[1];
  const float* v   = (const float*)d_in[2];
  const float* rot = (const float*)d_in[3];
  float* out = (float*)d_out;

  char* ws = (char*)d_ws;
  uint8_t* q8  = (uint8_t*)(ws);
  uint8_t* k8  = (uint8_t*)(ws + (4u << 20));
  uint8_t* v8p = (uint8_t*)(ws + (6u << 20));
  float*   qsc = (float*)(ws + (8u << 20));
  float*   ksc = qsc + 256;
  float*   vsc = ksc + 64;

  rotquant_kernel<true ><<<dim3(16, 16), 256, 0, stream>>>(q, rot, q8, qsc,
                                                           2048, 128, 16);
  rotquant_kernel<true ><<<dim3(8, 8),   256, 0, stream>>>(k, rot, k8, ksc,
                                                           2048, 256, 8);
  rotquant_kernel<false><<<dim3(8, 8),   256, 0, stream>>>(v, nullptr, v8p, vsc,
                                                           2048, 256, 8);
  fa_fp8_kernel<<<dim3(16, 16), 256, 0, stream>>>(q8, k8, v8p, qsc, ksc, vsc,
                                                  out);
}
